// LocalAggregator_30897994728148
// MI455X (gfx1250) — matrix operand order repaired and checked by reading, not compiled, not run on hardware
//
#include <hip/hip_runtime.h>

typedef __attribute__((ext_vector_type(16))) _Float16 v16h;
typedef __attribute__((ext_vector_type(2)))  __fp16   v2fp16;  // cvt_pkrtz result type
typedef __attribute__((ext_vector_type(8)))  float    v8f;
typedef __attribute__((ext_vector_type(4)))  unsigned int v4u;

#define NPTS   16384
#define NGAUSS 4096
#define NCLS   18
#define CHUNK  128   // gaussians staged per block iteration
#define SEMP   (CHUNK + 8)  // padded row: 272B stride -> conflict-free column reads
#define WAVES  8     // waves per block (256 threads, wave32)

__global__ __launch_bounds__(256)
void LocalAggregator_30897994728148_kernel(
    const float* __restrict__ pts,      // [NPTS,3]
    const float* __restrict__ means,    // [NGAUSS,3]
    const float* __restrict__ opac,     // [NGAUSS]
    const float* __restrict__ sem,      // [NGAUSS,NCLS]
    const float* __restrict__ scales,   // [NGAUSS,3]
    const float* __restrict__ cov,      // [NGAUSS,3,3]
    float* __restrict__ out)            // [NPTS,NCLS]
{
    // 16-byte gaussian records (3x ds_load_b128 per weight, broadcast per half-wave)
    __shared__ __align__(16) float4 s_gA[CHUNK];  // {mux, muy, muz, opa}
    __shared__ __align__(16) float4 s_gB[CHUNK];  // {-.5cxx, -.5cyy, -.5czz, -cxy}
    __shared__ __align__(16) float4 s_gC[CHUNK];  // {-cyz, -cxz, u32(mix|miy<<8|miz<<16|rad<<24), 0}
    // semantics transposed: [class][gaussian], f16, padded rows
    __shared__ __align__(16) _Float16 s_semT[32][SEMP];

    const int tid  = threadIdx.x;
    const int lane = tid & 31;
    const int wv   = tid >> 5;
    const int half = lane >> 4;   // 0: lanes 0-15, 1: lanes 16-31
    const int col  = lane & 15;

    // point tile owned by this wave; A rows: both half-waves hold M=0..15
    const int tile = blockIdx.x * WAVES + wv;
    const int prow = tile * 16 + col;
    const float px = pts[prow * 3 + 0];
    const float py = pts[prow * 3 + 1];
    const float pz = pts[prow * 3 + 2];
    const int pix = (int)((px + 40.0f) * 2.5f);
    const int piy = (int)((py + 40.0f) * 2.5f);
    const int piz = (int)((pz +  1.0f) * 2.5f);

    v8f c0 = {};   // classes  0..15
    v8f c1 = {};   // classes 16..31 (only 16,17 real)

    for (int cb = 0; cb < NGAUSS; cb += CHUNK) {
        __syncthreads();
        if (tid < CHUNK) {
            const int g = cb + tid;
            const float mx = means[g * 3 + 0];
            const float my = means[g * 3 + 1];
            const float mz = means[g * 3 + 2];
            s_gA[tid] = make_float4(mx, my, mz, opac[g]);
            s_gB[tid] = make_float4(-0.5f * cov[g * 9 + 0],
                                    -0.5f * cov[g * 9 + 4],
                                    -0.5f * cov[g * 9 + 8],
                                    -cov[g * 9 + 1]);
            const unsigned mix = (unsigned)(int)((mx + 40.0f) * 2.5f);
            const unsigned miy = (unsigned)(int)((my + 40.0f) * 2.5f);
            const unsigned miz = (unsigned)(int)((mz +  1.0f) * 2.5f);
            const float smax = fmaxf(scales[g * 3 + 0],
                               fmaxf(scales[g * 3 + 1], scales[g * 3 + 2]));
            const unsigned rad = (unsigned)(int)ceilf(smax * 7.5f);
            const unsigned pk  = mix | (miy << 8) | (miz << 16) | (rad << 24);
            s_gC[tid] = make_float4(-cov[g * 9 + 5], -cov[g * 9 + 2],
                                    __uint_as_float(pk), 0.0f);
        }
        for (int i = tid; i < CHUNK * 32; i += 256) {
            const int g = i >> 5, c = i & 31;
            const float v = (c < NCLS) ? sem[(cb + g) * NCLS + c] : 0.0f;
            s_semT[c][g] = (_Float16)v;
        }
        __syncthreads();

        #pragma unroll
        for (int kb = 0; kb < CHUNK; kb += 32) {
            // B tiles: K = (e&7) + 16*(e>>3) + 8*half, 8-element runs -> vector LDS loads
            union { v16h h; v4u u[2]; } b0, b1;
            const int kbB = kb + (half << 3);
            b0.u[0] = *(const v4u*)&s_semT[col][kbB];
            b0.u[1] = *(const v4u*)&s_semT[col][kbB + 16];
            b1.u[0] = *(const v4u*)&s_semT[16 + col][kbB];
            b1.u[1] = *(const v4u*)&s_semT[16 + col][kbB + 16];

            // A tile: lanes 0-15 K = e<8?e:e+8 ; lanes 16-31: +8 (pairs contiguous)
            union { v16h h; v2fp16 p[8]; } a;
            #pragma unroll
            for (int e = 0; e < 16; e += 2) {
                float w01[2];
                #pragma unroll
                for (int j = 0; j < 2; ++j) {
                    const int ee = e + j;
                    const int kA = kb + (ee & 7) + ((ee >> 3) << 4) + (half << 3);
                    const float4 gA = s_gA[kA];
                    const float4 gB = s_gB[kA];
                    const float4 gC = s_gC[kA];
                    const float dx = px - gA.x;
                    const float dy = py - gA.y;
                    const float dz = pz - gA.z;
                    // d^T (-0.5 * Sigma^{-1}) d as FMA chain
                    const float t  = fmaf(gB.x, dx, fmaf(gB.w, dy, gC.y * dz));
                    const float u  = fmaf(gB.y, dy, gC.x * dz);
                    const float pw = fmaf(dx, t, fmaf(dy, u, gB.z * (dz * dz)));
                    const unsigned pk = __float_as_uint(gC.z);
                    const int rad = (int)(pk >> 24);
                    const bool within = (abs(pix - (int)(pk & 255u))         <= rad) &
                                        (abs(piy - (int)((pk >> 8)  & 255u)) <= rad) &
                                        (abs(piz - (int)((pk >> 16) & 255u)) <= rad);
                    w01[j] = within ? gA.w * __expf(pw) : 0.0f;
                }
                a.p[e >> 1] = __builtin_amdgcn_cvt_pkrtz(w01[0], w01[1]);
            }

            c0 = __builtin_amdgcn_wmma_f32_16x16x32_f16(
                     false, a.h, false, b0.h, (short)0, c0, false, false);
            c1 = __builtin_amdgcn_wmma_f32_16x16x32_f16(
                     false, a.h, false, b1.h, (short)0, c1, false, false);
        }
    }

    // D layout (f32 16x16): VGPR r -> M = half*8 + r, N = col
    const int base = tile * 16;
    #pragma unroll
    for (int r = 0; r < 8; ++r) {
        const int row = base + half * 8 + r;
        out[row * NCLS + col] = c0[r];
        if (col < 2) out[row * NCLS + 16 + col] = c1[r];
    }
}

extern "C" void kernel_launch(void* const* d_in, const int* in_sizes, int n_in,
                              void* d_out, int out_size, void* d_ws, size_t ws_size,
                              hipStream_t stream) {
    (void)in_sizes; (void)n_in; (void)out_size; (void)d_ws; (void)ws_size;
    const float* pts    = (const float*)d_in[0];
    const float* means  = (const float*)d_in[1];
    const float* opac   = (const float*)d_in[2];
    const float* sem    = (const float*)d_in[3];
    const float* scales = (const float*)d_in[4];
    const float* cov    = (const float*)d_in[5];
    float* out = (float*)d_out;

    const int blocks = NPTS / (16 * WAVES);  // 128
    LocalAggregator_30897994728148_kernel<<<blocks, 256, 0, stream>>>(
        pts, means, opac, sem, scales, cov, out);
}
